// SNNetwork_14310831030717
// MI455X (gfx1250) — matrix operand order repaired and checked by reading, not compiled, not run on hardware
//
#include <hip/hip_runtime.h>
#include <stdint.h>

// Problem constants
#define HID     4096
#define KDIM    2048
#define NC      12
#define NSTEPS  15
#define BATCH   64
#define NWIN    3
#define NGROUP  (BATCH * NWIN)     // 192 LIF groups (batch x window)
#define MROWS   (NGROUP * 16)      // 3072: each group padded 15 -> 16 rows

typedef __bf16        bf16x16 __attribute__((ext_vector_type(16)));
typedef float         f32x8   __attribute__((ext_vector_type(8)));
typedef unsigned int  u32x4   __attribute__((ext_vector_type(4)));

union FragAB { u32x4 u[2]; bf16x16 b; };

__device__ __forceinline__ unsigned short f32_to_bf16(float f) {
    unsigned int u = __float_as_uint(f);
    unsigned int r = u + 0x7FFFu + ((u >> 16) & 1u);   // round-to-nearest-even
    return (unsigned short)(r >> 16);
}

// ---------------------------------------------------------------------------
// K0: pack w1 [2048, 4096] f32  ->  Bp [4096][2048] bf16 (transposed, n-major)
// B-fragment layout wants k-pairs contiguous per output column n.
// ---------------------------------------------------------------------------
__global__ void pack_w1_kernel(const float* __restrict__ w1,
                               unsigned short* __restrict__ Bp) {
    __shared__ unsigned short tile[32][33];
    int k0 = blockIdx.x * 32;
    int n0 = blockIdx.y * 32;
    for (int i = threadIdx.y; i < 32; i += 8)
        tile[i][threadIdx.x] =
            f32_to_bf16(w1[(size_t)(k0 + i) * HID + n0 + threadIdx.x]);
    __syncthreads();
    for (int i = threadIdx.y; i < 32; i += 8)
        Bp[(size_t)(n0 + i) * KDIM + k0 + threadIdx.x] = tile[threadIdx.x][i];
}

// ---------------------------------------------------------------------------
// K1: fuse 1x1 antenna conv and pack A:
//   A[row = g*16 + s][d] = bf16( sum_c x[b, w*15+s, d, c] * cw[c] + cb )
// where g = b*3 + w; row s==15 is a zero pad so each LIF group = 16 rows.
// ---------------------------------------------------------------------------
__global__ void conv_pack_kernel(const float* __restrict__ x,
                                 const float* __restrict__ cw,
                                 const float* __restrict__ cb,
                                 unsigned short* __restrict__ Ap) {
    int row = blockIdx.x;             // 0..3071
    int g = row >> 4;
    int s = row & 15;
    unsigned short* dst = Ap + (size_t)row * KDIM;
    if (s == NSTEPS) {                // pad row
        for (int d = threadIdx.x; d < KDIM; d += blockDim.x) dst[d] = 0;
        return;
    }
    int b = g / NWIN, w = g % NWIN;
    int t = w * NSTEPS + s;
    const float4* xr = (const float4*)(x) + (size_t)(b * 45 + t) * KDIM;
    float c0 = cw[0], c1 = cw[1], c2 = cw[2], c3 = cw[3], cb0 = cb[0];
    for (int d = threadIdx.x; d < KDIM; d += blockDim.x) {
        float4 v = xr[d];
        float h = v.x * c0 + v.y * c1 + v.z * c2 + v.w * c3 + cb0;
        dst[d] = f32_to_bf16(h);
    }
}

// ---------------------------------------------------------------------------
// K2: bf16 WMMA GEMM [3072 x 2048] @ [2048 x 4096] with fused bias + LIF scan
// over the 15 step rows of each 16-row group; writes spike-sum [192, 4096].
// Workgroup: 8 waves (2 x 4), tile 128 rows x 256 cols; wave tile 64x64.
// Fragments loaded straight from the packed layouts as 2x b128 per lane.
// ---------------------------------------------------------------------------
__global__ __launch_bounds__(256)
void gemm_lif_kernel(const unsigned int* __restrict__ Apu,   // A bf16 pairs
                     const unsigned int* __restrict__ Bpu,   // Bp bf16 pairs
                     const float* __restrict__ b1,
                     const float* __restrict__ beta1p,
                     const float* __restrict__ thr1p,
                     float* __restrict__ spksum) {
    const int lane  = threadIdx.x & 31;
    const int wave  = threadIdx.x >> 5;
    const int half  = lane >> 4;        // lane 16..31 -> upper K half
    const int laneM = lane & 15;        // row (A) / col (B) within fragment
    const int wm    = wave >> 2;        // 0..1
    const int wn    = wave & 3;         // 0..3

    const int mBase = blockIdx.y * 128 + wm * 64;
    const int nBase = blockIdx.x * 256 + wn * 64;

    const f32x8 vzero = {0.f, 0.f, 0.f, 0.f, 0.f, 0.f, 0.f, 0.f};
    f32x8 acc[4][4];
#pragma unroll
    for (int i = 0; i < 4; i++)
#pragma unroll
        for (int j = 0; j < 4; j++) acc[i][j] = vzero;

    const int KPAIR = KDIM / 2;         // 1024 uints per row/col
    const unsigned int* aRow[4];
#pragma unroll
    for (int mf = 0; mf < 4; mf++)      // A: regs0-3 k=2r+8h, regs4-7 +16
        aRow[mf] = Apu + (size_t)(mBase + mf * 16 + laneM) * KPAIR + 4 * half;
    const unsigned int* bCol[4];
#pragma unroll
    for (int nf = 0; nf < 4; nf++)      // B: regs0-3 k=2r+8h, regs4-7 +16
        bCol[nf] = Bpu + (size_t)(nBase + nf * 16 + laneM) * KPAIR + 4 * half;

    for (int kc = 0; kc < KDIM / 32; kc++) {
        FragAB fa[4];
#pragma unroll
        for (int mf = 0; mf < 4; mf++) {
            fa[mf].u[0] = *(const u32x4*)(aRow[mf] + kc * 16);
            fa[mf].u[1] = *(const u32x4*)(aRow[mf] + kc * 16 + 8);
        }
#pragma unroll
        for (int nf = 0; nf < 4; nf++) {
            FragAB fb;
            fb.u[0] = *(const u32x4*)(bCol[nf] + kc * 16);
            fb.u[1] = *(const u32x4*)(bCol[nf] + kc * 16 + 8);
#pragma unroll
            for (int mf = 0; mf < 4; mf++) {
                acc[mf][nf] = __builtin_amdgcn_wmma_f32_16x16x32_bf16(
                    false, fa[mf].b, false, fb.b, (short)0, acc[mf][nf],
                    false, false);
            }
        }
    }

    // ---- Epilogue: per 16-row group, bias + LIF over 15 steps, spike-sum ----
    __shared__ float buf[8][64][17];    // [wave][col][row], padded vs conflicts
    const float beta = fminf(fmaxf(beta1p[0], 0.f), 1.f);
    const float thr  = thr1p[0];

    for (int mf = 0; mf < 4; mf++) {
#pragma unroll
        for (int nf = 0; nf < 4; nf++)
#pragma unroll
            for (int r = 0; r < 8; r++)   // C layout: VGPR r -> row r + 8*half
                buf[wave][nf * 16 + laneM][r + 8 * half] = acc[mf][nf][r];
        __syncthreads();

        int gg = (mBase + mf * 16) >> 4;              // LIF group index
        for (int cc = lane; cc < 64; cc += 32) {
            int n = nBase + cc;
            float bias = b1[n];
            float mem = 0.f, ssum = 0.f;
#pragma unroll
            for (int st = 0; st < NSTEPS; st++) {     // pad row 15 ignored
                float inp     = buf[wave][cc][st] + bias;
                float reset   = (mem - thr) > 0.f ? 1.f : 0.f;
                float mem_pre = beta * mem + inp - reset * thr;
                float spk     = (mem_pre - thr) > 0.f ? 1.f : 0.f;
                mem  = mem_pre - (spk - reset) * thr;
                ssum += spk;
            }
            spksum[(size_t)gg * HID + n] = ssum;
        }
        __syncthreads();
    }
}

// ---------------------------------------------------------------------------
// K3: hid[192,12] = spksum[192,4096] @ w2[4096,12] + b2  (bandwidth trivial)
// ---------------------------------------------------------------------------
__global__ void gemm2_kernel(const float* __restrict__ spksum,
                             const float* __restrict__ w2,
                             const float* __restrict__ b2,
                             float* __restrict__ hid) {
    int row = blockIdx.x;
    int t = threadIdx.x;
    float acc[NC];
#pragma unroll
    for (int c = 0; c < NC; c++) acc[c] = 0.f;
    for (int k = t; k < HID; k += 256) {
        float s = spksum[(size_t)row * HID + k];
        const float* wr = w2 + (size_t)k * NC;
#pragma unroll
        for (int c = 0; c < NC; c++) acc[c] += s * wr[c];
    }
    __shared__ float sm[256];
    for (int c = 0; c < NC; c++) {
        sm[t] = acc[c];
        __syncthreads();
        for (int off = 128; off > 0; off >>= 1) {
            if (t < off) sm[t] += sm[t + off];
            __syncthreads();
        }
        if (t == 0) hid[(size_t)row * NC + c] = sm[0] + b2[c];
        __syncthreads();
    }
}

// ---------------------------------------------------------------------------
// K4: LIF over the 3 windows, spike-sum, softmax over 12 classes.
// ---------------------------------------------------------------------------
__global__ void lif2_softmax_kernel(const float* __restrict__ hid,
                                    const float* __restrict__ beta2p,
                                    const float* __restrict__ thr2p,
                                    float* __restrict__ out) {
    int b = threadIdx.x;
    if (b >= BATCH) return;
    float beta = fminf(fmaxf(beta2p[0], 0.f), 1.f);
    float thr  = thr2p[0];
    float mem[NC], ssum[NC];
#pragma unroll
    for (int c = 0; c < NC; c++) { mem[c] = 0.f; ssum[c] = 0.f; }
    for (int w = 0; w < NWIN; w++) {
#pragma unroll
        for (int c = 0; c < NC; c++) {
            float inp     = hid[(size_t)(b * NWIN + w) * NC + c];
            float reset   = (mem[c] - thr) > 0.f ? 1.f : 0.f;
            float mem_pre = beta * mem[c] + inp - reset * thr;
            float spk     = (mem_pre - thr) > 0.f ? 1.f : 0.f;
            mem[c] = mem_pre - (spk - reset) * thr;
            ssum[c] += spk;
        }
    }
    float mx = ssum[0];
#pragma unroll
    for (int c = 1; c < NC; c++) mx = fmaxf(mx, ssum[c]);
    float e[NC], den = 0.f;
#pragma unroll
    for (int c = 0; c < NC; c++) { e[c] = __expf(ssum[c] - mx); den += e[c]; }
    float inv = 1.f / den;
#pragma unroll
    for (int c = 0; c < NC; c++) out[(size_t)b * NC + c] = e[c] * inv;
}

// ---------------------------------------------------------------------------
extern "C" void kernel_launch(void* const* d_in, const int* in_sizes, int n_in,
                              void* d_out, int out_size, void* d_ws,
                              size_t ws_size, hipStream_t stream) {
    const float* x     = (const float*)d_in[0];
    const float* cw    = (const float*)d_in[1];
    const float* cb    = (const float*)d_in[2];
    const float* w1    = (const float*)d_in[3];
    const float* b1    = (const float*)d_in[4];
    const float* w2    = (const float*)d_in[5];
    const float* b2    = (const float*)d_in[6];
    const float* beta1 = (const float*)d_in[7];
    const float* thr1  = (const float*)d_in[8];
    const float* beta2 = (const float*)d_in[9];
    const float* thr2  = (const float*)d_in[10];
    float* out = (float*)d_out;

    char* ws = (char*)d_ws;
    size_t offA = 0;                                         // A bf16: 12.6 MB
    size_t offB = offA + (size_t)MROWS * KDIM * 2;           // Bp bf16: 16.8 MB
    size_t offS = offB + (size_t)HID * KDIM * 2;             // spksum:   3.1 MB
    size_t offH = offS + (size_t)NGROUP * HID * 4;           // hid:      9 KB
    unsigned short* Ap     = (unsigned short*)(ws + offA);
    unsigned short* Bp     = (unsigned short*)(ws + offB);
    float*          spksum = (float*)(ws + offS);
    float*          hid    = (float*)(ws + offH);

    // K0: transpose-pack w1 to bf16
    pack_w1_kernel<<<dim3(KDIM / 32, HID / 32), dim3(32, 8), 0, stream>>>(w1, Bp);
    // K1: fused antenna conv + A pack (with per-group pad row)
    conv_pack_kernel<<<MROWS, 256, 0, stream>>>(x, cw, cb, Ap);
    // K2: WMMA GEMM + bias + LIF(15 steps) + spike-sum
    gemm_lif_kernel<<<dim3(HID / 256, MROWS / 128), 256, 0, stream>>>(
        (const unsigned int*)Ap, (const unsigned int*)Bp, b1, beta1, thr1,
        spksum);
    // K3: small classifier GEMM
    gemm2_kernel<<<NGROUP, 256, 0, stream>>>(spksum, w2, b2, hid);
    // K4: window LIF + softmax
    lif2_softmax_kernel<<<1, 64, 0, stream>>>(hid, beta2, thr2, out);
}
